// LlamaAttention_27290222199359
// MI455X (gfx1250) — matrix operand order repaired and checked by reading, not compiled, not run on hardware
//
#include <hip/hip_runtime.h>

typedef __attribute__((ext_vector_type(16))) __bf16 v16bf;
typedef __attribute__((ext_vector_type(8)))  __bf16 v8bf;
typedef __attribute__((ext_vector_type(8)))  float  v8f;

#define S_LEN 2048
#define HID   2048
#define NH    16
#define HD    128
#define BATCH 2
#define MTOT  (BATCH * S_LEN)   // 4096 tokens

// ---------------------------------------------------------------------------
// CDNA5 async global -> LDS copy (16B per lane), tracked by ASYNCcnt.
// ---------------------------------------------------------------------------
__device__ __forceinline__ void async_copy_b128(unsigned int lds_off, const __bf16* gp) {
  asm volatile("global_load_async_to_lds_b128 %0, %1, off"
               :: "v"(lds_off), "v"(gp)
               : "memory");
}
__device__ __forceinline__ void wait_async0() {
  asm volatile("s_wait_asynccnt 0x0" ::: "memory");
}
__device__ __forceinline__ unsigned int lds_addr32(const void* p) {
  return (unsigned int)(unsigned long long)p;  // flat LDS aperture: low 32 bits = LDS offset
}

// ---------------------------------------------------------------------------
// fp32 -> bf16 elementwise convert
// ---------------------------------------------------------------------------
__global__ void cvt_bf16(const float* __restrict__ in, __bf16* __restrict__ out, int n) {
  int i = blockIdx.x * blockDim.x + threadIdx.x;
  int stride = gridDim.x * blockDim.x;
  for (; i < n; i += stride) out[i] = (__bf16)in[i];
}

// ---------------------------------------------------------------------------
// NT GEMM: C[m,n] = sum_k A[m,k] * Bw[n,k]   (A: M x K, Bw: N x K, row-major)
// M=4096, N=2048, K=2048. Block tile 128x128, 8 waves (wave32), K-step 64.
// Double-buffered LDS filled by async global->LDS DMA; 1 barrier per stage.
// Inner slab: load all 8 B fragments, then 8 back-to-back WMMAs (staggered
// dscnt waits -> LDS/XDL overlap within a wave).
// mode 0: fp32 store row-major [M,N]                (final out projection)
// mode 1: RoPE in registers, bf16 store [b,h,s,d]   (Q, K)
// mode 2: bf16 store transposed [b,h,d,s]           (V)
// ---------------------------------------------------------------------------
#define KSTEP 64
#define LROW  72   // 64 + 8 pad elements: 144B row stride, conflict-free b128 reads

__launch_bounds__(256)
__global__ void gemm_nt(const __bf16* __restrict__ A, const __bf16* __restrict__ Bw,
                        void* __restrict__ out, int mode) {
  const int K = HID, N = HID;
  __shared__ __align__(16) __bf16 As[2][128][LROW];
  __shared__ __align__(16) __bf16 Bs[2][128][LROW];

  int tid  = threadIdx.x;
  int wave = tid >> 5, L = tid & 31, l15 = L & 15, hiw = L >> 4;
  int m0 = blockIdx.y * 128, n0 = blockIdx.x * 128;

  v8f acc[8];
  for (int i = 0; i < 8; i++)
    for (int j = 0; j < 8; j++) acc[i][j] = 0.0f;

  // staging assignment: each thread owns 32 contiguous bf16 of one row of A and B
  int row = tid >> 1, half = tid & 1;
  const __bf16* Arow = A  + (size_t)(m0 + row) * K + half * 32;
  const __bf16* Brow = Bw + (size_t)(n0 + row) * K + half * 32;
  int colbase = half * 32;

  // prologue: fill buffer 0
  for (int c = 0; c < 4; c++) {
    async_copy_b128(lds_addr32(&As[0][row][colbase + c * 8]), Arow + 0 + c * 8);
    async_copy_b128(lds_addr32(&Bs[0][row][colbase + c * 8]), Brow + 0 + c * 8);
  }

  int stage = 0;
  for (int kt = 0; kt < K; kt += KSTEP, stage ^= 1) {
    wait_async0();        // my DMA into buffer `stage` has landed
    __syncthreads();      // everyone's has; also fences prior reads of stage^1
    if (kt + KSTEP < K) {
      for (int c = 0; c < 4; c++) {
        async_copy_b128(lds_addr32(&As[stage ^ 1][row][colbase + c * 8]),
                        Arow + (kt + KSTEP) + c * 8);
        async_copy_b128(lds_addr32(&Bs[stage ^ 1][row][colbase + c * 8]),
                        Brow + (kt + KSTEP) + c * 8);
      }
    }

    // compute on buffer `stage`: 2 x (K=32) slabs, 8 N-subtiles each
    for (int kk = 0; kk < KSTEP; kk += 32) {
      int am = wave * 16 + l15;
      int koff = kk + hiw * 8;
      v8bf alo = *(const v8bf*)&As[stage][am][koff];
      v8bf ahi = *(const v8bf*)&As[stage][am][koff + 16];
      v16bf afrag;
      for (int i = 0; i < 8; i++) { afrag[i] = alo[i]; afrag[8 + i] = ahi[i]; }

      // load all 8 B fragments first so ds_loads stay in flight across WMMAs
      v16bf bf[8];
      for (int ns = 0; ns < 8; ns++) {
        int bn = ns * 16 + l15;
        int k2 = kk + hiw * 8;
        v8bf blo = *(const v8bf*)&Bs[stage][bn][k2];
        v8bf bhi = *(const v8bf*)&Bs[stage][bn][k2 + 16];
        for (int i = 0; i < 8; i++) { bf[ns][i] = blo[i]; bf[ns][8 + i] = bhi[i]; }
      }
      for (int ns = 0; ns < 8; ns++) {
        acc[ns] = __builtin_amdgcn_wmma_f32_16x16x32_bf16(
            false, afrag, false, bf[ns], (short)0, acc[ns], false, false);
      }
    }
  }

  // C layout: VGPR r, lanes 0-15 -> (M=r, N=lane); lanes 16-31 -> (M=r+8, N=lane-16)
  int mbase = m0 + wave * 16 + hiw * 8;

  if (mode == 0) {
    float* C = (float*)out;
    for (int ns = 0; ns < 8; ns++) {
      int n = n0 + ns * 16 + l15;
      for (int r = 0; r < 8; r++)
        C[(size_t)(mbase + r) * N + n] = acc[ns][r];
    }
  } else if (mode == 1) {
    // fused RoPE: pairs (d, d+64) are acc[ns] / acc[ns+4] in the same lane
    __bf16* O = (__bf16*)out;
    int h = n0 >> 7;  // tile N==128 aligns exactly with one head
    for (int ns = 0; ns < 4; ns++) {
      int d = ns * 16 + l15;  // 0..63
      float inv = __expf(-(float)d * (9.210340371976184f / 64.0f));  // 10000^(-d/64)
      for (int r = 0; r < 8; r++) {
        int m = mbase + r;
        int b = m >> 11, s = m & (S_LEN - 1);
        float sn, cs;
        __sincosf((float)s * inv, &sn, &cs);
        float x1 = acc[ns][r], x2 = acc[ns + 4][r];
        size_t base = ((size_t)(b * NH + h) * S_LEN + s) * HD;
        O[base + d]      = (__bf16)(x1 * cs - x2 * sn);
        O[base + d + 64] = (__bf16)(x2 * cs + x1 * sn);
      }
    }
  } else {
    // V transposed store [b,h,d,s] so attention's B operand is contiguous
    __bf16* O = (__bf16*)out;
    int h = n0 >> 7;
    for (int ns = 0; ns < 8; ns++) {
      int d = ns * 16 + l15;
      for (int r = 0; r < 8; r++) {
        int m = mbase + r;
        int b = m >> 11, s = m & (S_LEN - 1);
        O[((size_t)(b * NH + h) * HD + d) * S_LEN + s] = (__bf16)acc[ns][r];
      }
    }
  }
}

// ---------------------------------------------------------------------------
// Flash attention (causal), bf16 WMMA, fp32 online softmax.
// Grid: (S/128, NH, B), 256 threads = 8 waves; each wave owns 16 q-rows.
// Q,K: [b,h,s,d] bf16 (RoPE applied). V: [b,h,d,s] bf16. Out: [b,s,h*d] bf16.
// ---------------------------------------------------------------------------
__launch_bounds__(256)
__global__ void flash_attn(const __bf16* __restrict__ Qh, const __bf16* __restrict__ Kh,
                           const __bf16* __restrict__ Vt, __bf16* __restrict__ Obf) {
  __shared__ __align__(16) __bf16 Ps[8][16][40];  // per-wave P tile (16 q x 32 kv)

  int tid = threadIdx.x, wave = tid >> 5, L = tid & 31, l15 = L & 15, hiw = L >> 4;
  int b = blockIdx.z, h = blockIdx.y;
  int qbase = blockIdx.x * 128 + wave * 16;
  const float scale = 0.08838834764831845f;  // 1/sqrt(128)

  const __bf16* Qb = Qh + (size_t)(b * NH + h) * S_LEN * HD;
  const __bf16* Kb = Kh + (size_t)(b * NH + h) * S_LEN * HD;
  const __bf16* Vb = Vt + (size_t)(b * NH + h) * HD * S_LEN;

  // Q fragments (A operand), 4 chunks of K=32 over d=128
  v16bf qf[4];
  {
    int s = qbase + l15;
    int koff = hiw * 8;
    for (int dc = 0; dc < 4; dc++) {
      const __bf16* p = Qb + (size_t)s * HD + dc * 32 + koff;
      v8bf lo = *(const v8bf*)p;
      v8bf hi8 = *(const v8bf*)(p + 16);
      for (int i = 0; i < 8; i++) { qf[dc][i] = lo[i]; qf[dc][8 + i] = hi8[i]; }
    }
  }

  float mi[8], li[8];
  v8f acc[8];
  for (int r = 0; r < 8; r++) { mi[r] = -1e30f; li[r] = 0.0f; }
  for (int d8 = 0; d8 < 8; d8++)
    for (int r = 0; r < 8; r++) acc[d8][r] = 0.0f;

  int qmax = qbase + 15;
  int nch = qmax / 32 + 1;  // causal: skip fully-masked kv chunks

  for (int kc = 0; kc < nch; kc++) {
    int kvb = kc * 32;
    v8f sc0, sc1;
    for (int r = 0; r < 8; r++) { sc0[r] = 0.0f; sc1[r] = 0.0f; }

    // K fragments for both kv halves, all d chunks, loaded up front
    v16bf kf0[4], kf1[4];
    for (int dc = 0; dc < 4; dc++) {
      {
        int kv = kvb + l15;
        const __bf16* p = Kb + (size_t)kv * HD + dc * 32 + hiw * 8;
        v8bf lo = *(const v8bf*)p; v8bf h8 = *(const v8bf*)(p + 16);
        for (int i = 0; i < 8; i++) { kf0[dc][i] = lo[i]; kf0[dc][8 + i] = h8[i]; }
      }
      {
        int kv = kvb + 16 + l15;
        const __bf16* p = Kb + (size_t)kv * HD + dc * 32 + hiw * 8;
        v8bf lo = *(const v8bf*)p; v8bf h8 = *(const v8bf*)(p + 16);
        for (int i = 0; i < 8; i++) { kf1[dc][i] = lo[i]; kf1[dc][8 + i] = h8[i]; }
      }
    }
    for (int dc = 0; dc < 4; dc++) {
      sc0 = __builtin_amdgcn_wmma_f32_16x16x32_bf16(
          false, qf[dc], false, kf0[dc], (short)0, sc0, false, false);
      sc1 = __builtin_amdgcn_wmma_f32_16x16x32_bf16(
          false, qf[dc], false, kf1[dc], (short)0, sc1, false, false);
    }

    // online softmax; row M lives across one 16-lane half -> xor-shuffle reduce
    float p0a[8], p1a[8], fscale[8];
    for (int r = 0; r < 8; r++) {
      int q = qbase + r + hiw * 8;
      int k0 = kvb + l15, k1 = kvb + 16 + l15;
      float s0 = sc0[r] * scale + (k0 > q ? -1e9f : 0.0f);
      float s1 = sc1[r] * scale + (k1 > q ? -1e9f : 0.0f);
      float mx = fmaxf(s0, s1);
      for (int off = 1; off < 16; off <<= 1) mx = fmaxf(mx, __shfl_xor(mx, off, 32));
      float mnew = fmaxf(mi[r], mx);
      float f  = __expf(mi[r] - mnew);
      float p0 = __expf(s0 - mnew);
      float p1 = __expf(s1 - mnew);
      float rs = p0 + p1;
      for (int off = 1; off < 16; off <<= 1) rs += __shfl_xor(rs, off, 32);
      li[r] = li[r] * f + rs;
      mi[r] = mnew;
      p0a[r] = p0; p1a[r] = p1; fscale[r] = f;
    }
    for (int d8 = 0; d8 < 8; d8++)
      for (int r = 0; r < 8; r++) acc[d8][r] *= fscale[r];

    // P: C layout -> A layout via per-wave LDS bounce
    for (int r = 0; r < 8; r++) {
      int M = r + hiw * 8;
      Ps[wave][M][l15]      = (__bf16)p0a[r];
      Ps[wave][M][16 + l15] = (__bf16)p1a[r];
    }
    __builtin_amdgcn_wave_barrier();
    v16bf pf;
    {
      int koff = hiw * 8;
      v8bf lo = *(const v8bf*)&Ps[wave][l15][koff];
      v8bf h8 = *(const v8bf*)&Ps[wave][l15][koff + 16];
      for (int i = 0; i < 8; i++) { pf[i] = lo[i]; pf[8 + i] = h8[i]; }
    }
    __builtin_amdgcn_wave_barrier();

    // O += P @ V : 8 d-subtiles; load all V fragments, then back-to-back WMMAs
    v16bf vf[8];
    for (int d8 = 0; d8 < 8; d8++) {
      int d = d8 * 16 + l15;
      const __bf16* p = Vb + (size_t)d * S_LEN + kvb + hiw * 8;
      v8bf lo = *(const v8bf*)p; v8bf h8 = *(const v8bf*)(p + 16);
      for (int i = 0; i < 8; i++) { vf[d8][i] = lo[i]; vf[d8][8 + i] = h8[i]; }
    }
    for (int d8 = 0; d8 < 8; d8++) {
      acc[d8] = __builtin_amdgcn_wmma_f32_16x16x32_bf16(
          false, pf, false, vf[d8], (short)0, acc[d8], false, false);
    }
  }

  // epilogue: O / l  -> bf16 [b, s, h*128+d] (row-major for the final GEMM)
  for (int d8 = 0; d8 < 8; d8++) {
    int d = d8 * 16 + l15;
    for (int r = 0; r < 8; r++) {
      int q = qbase + r + hiw * 8;
      float val = acc[d8][r] / li[r];
      Obf[((size_t)(b * S_LEN) + q) * HID + h * HD + d] = (__bf16)val;
    }
  }
}

// ---------------------------------------------------------------------------
// launcher
// ---------------------------------------------------------------------------
extern "C" void kernel_launch(void* const* d_in, const int* in_sizes, int n_in,
                              void* d_out, int out_size, void* d_ws, size_t ws_size,
                              hipStream_t stream) {
  (void)in_sizes; (void)n_in; (void)out_size; (void)ws_size;
  const float* X  = (const float*)d_in[0];
  // d_in[1] = attention_mask (causal, reproduced analytically)
  const float* Wq = (const float*)d_in[2];
  const float* Wk = (const float*)d_in[3];
  const float* Wv = (const float*)d_in[4];
  const float* Wo = (const float*)d_in[5];
  // d_in[6] = position_ids (= arange, reproduced analytically)

  char* ws = (char*)d_ws;
  size_t off = 0;
  auto alloc = [&](size_t bytes) -> void* {
    void* p = ws + off;
    off += (bytes + 255) & ~(size_t)255;
    return p;
  };

  const size_t xBytes = (size_t)MTOT * HID * sizeof(__bf16);
  const size_t wBytes = (size_t)HID * HID * sizeof(__bf16);
  __bf16* Xb  = (__bf16*)alloc(xBytes);
  __bf16* Wqb = (__bf16*)alloc(wBytes);
  __bf16* Wkb = (__bf16*)alloc(wBytes);
  __bf16* Wvb = (__bf16*)alloc(wBytes);
  __bf16* Wob = (__bf16*)alloc(wBytes);
  __bf16* Qh  = (__bf16*)alloc(xBytes);   // [b,h,s,d]
  __bf16* Kk  = (__bf16*)alloc(xBytes);   // [b,h,s,d]
  __bf16* Vt  = (__bf16*)alloc(xBytes);   // [b,h,d,s]
  __bf16* Ob  = (__bf16*)alloc(xBytes);   // [b,s,h*d]

  int nX = MTOT * HID;   // 8,388,608
  int nW = HID * HID;    // 4,194,304
  cvt_bf16<<<4096, 256, 0, stream>>>(X,  Xb,  nX);
  cvt_bf16<<<2048, 256, 0, stream>>>(Wq, Wqb, nW);
  cvt_bf16<<<2048, 256, 0, stream>>>(Wk, Wkb, nW);
  cvt_bf16<<<2048, 256, 0, stream>>>(Wv, Wvb, nW);
  cvt_bf16<<<2048, 256, 0, stream>>>(Wo, Wob, nW);

  dim3 gg(HID / 128, MTOT / 128);  // (16, 32)
  gemm_nt<<<gg, 256, 0, stream>>>(Xb, Wqb, Qh, 1);   // Q proj + RoPE
  gemm_nt<<<gg, 256, 0, stream>>>(Xb, Wkb, Kk, 1);   // K proj + RoPE
  gemm_nt<<<gg, 256, 0, stream>>>(Xb, Wvb, Vt, 2);   // V proj, transposed store

  dim3 ga(S_LEN / 128, NH, BATCH);  // (16, 16, 2)
  flash_attn<<<ga, 256, 0, stream>>>(Qh, Kk, Vt, Ob);

  gemm_nt<<<gg, 256, 0, stream>>>(Ob, Wob, d_out, 0);  // output projection
}
